// PureInt8Linear_44521630991151
// MI455X (gfx1250) — matrix operand order repaired and checked by reading, not compiled, not run on hardware
//
#include <hip/hip_runtime.h>
#include <stdint.h>

typedef _Float16 v16h __attribute__((ext_vector_type(16)));
typedef _Float16 v8h  __attribute__((ext_vector_type(8)));
typedef _Float16 v4h  __attribute__((ext_vector_type(4)));
typedef _Float16 v2h  __attribute__((ext_vector_type(2)));
typedef float    v8f  __attribute__((ext_vector_type(8)));
typedef int      v4i  __attribute__((ext_vector_type(4)));

// Problem constants (fixed by the reference).
#define KDIM   8192
#define NDIM   16384
#define BATCH  16
#define CHUNK  1024        // split-K chunk == quant block size -> 1 scale per lane
#define NSPLIT (KDIM / CHUNK)

// Block: 256 threads = 8 waves; each wave owns a 16-wide N tile -> 128 N per block.
// Grid: (NDIM/128, NSPLIT).
__global__ void __launch_bounds__(256)
int8_gemm_wmma_kernel(const float* __restrict__ x,
                      const signed char* __restrict__ qw,
                      const _Float16* __restrict__ scales,
                      float* __restrict__ ws) {
    __shared__ _Float16 lx[BATCH * CHUNK];   // 32 KB: x slice as fp16

    const int tid   = threadIdx.x;
    const int lane  = tid & 31;
    const int wave  = tid >> 5;
    const int kbase = blockIdx.y * CHUNK;

    // ---- cooperative load: x fp32 -> fp16 into LDS (coalesced float4 loads) ----
    // iteration i covers row m = i, thread t covers kk = 4t..4t+3
    #pragma unroll
    for (int i = 0; i < BATCH; ++i) {
        int idx4 = i * CHUNK + tid * 4;
        float4 v = *(const float4*)&x[(size_t)i * KDIM + kbase + tid * 4];
        v4h h = { (_Float16)v.x, (_Float16)v.y, (_Float16)v.z, (_Float16)v.w };
        *(v4h*)&lx[idx4] = h;                 // ds_store_b64, 8B aligned
    }
    __syncthreads();

    const int nBase = blockIdx.x * 128 + wave * 16;
    const int nCol  = nBase + (lane & 15);    // output column / weight row this lane feeds
    const int hi    = lane >> 4;              // 0: K 0..15 of step, 1: K 16..31

    // One per-lane dequant scale for the whole kernel (kchunk == quant block).
    const _Float16 sc = scales[(size_t)nCol * (KDIM / CHUNK) + blockIdx.y];
    const v2h sc2  = { sc, sc };
    const v2h off2 = { (_Float16)(-1152), (_Float16)(-1152) };  // exact in fp16

    // B stream: 16 contiguous int8 per lane per WMMA step.
    const signed char* wrow = qw + (size_t)nCol * KDIM + kbase + hi * 16;
    // A fragment source in LDS (contiguous K order: K = hi*16 + 0..15).
    const _Float16* arow = &lx[(lane & 15) * CHUNK + hi * 16];

    v8f acc = {};
    #pragma unroll 4
    for (int kk = 0; kk < CHUNK; kk += 32) {
        // A fragment: two 128-bit LDS reads -> 16 halves
        v8h alo = *(const v8h*)&arow[kk];          // K { hi*16 + 0..7 }
        v8h ahi = *(const v8h*)&arow[kk + 8];      // K { hi*16 + 8..15 }
        v16h a  = __builtin_shufflevector(alo, ahi,
                      0,1,2,3,4,5,6,7,8,9,10,11,12,13,14,15);

        // B fragment: 16 int8 (global_load_b128) -> fp16 via perm/xor bit trick.
        // fp16 bits (0x6400 | (b^0x80)) == 1024 + (b+128); subtract 1152 -> exact b;
        // multiply by scale -> RN(b*sc), identical to the reference dequant.
        v4i wq = *(const v4i*)&wrow[kk];
        v16h b;
        #pragma unroll
        for (int d = 0; d < 4; ++d) {
            unsigned w  = (unsigned)wq[d];
            unsigned p0 = __builtin_amdgcn_perm(0u, w, 0x0C010C00u) ^ 0x64806480u;
            unsigned p1 = __builtin_amdgcn_perm(0u, w, 0x0C030C02u) ^ 0x64806480u;
            v2h h0 = (__builtin_bit_cast(v2h, p0) + off2) * sc2;   // pk_add + pk_mul
            v2h h1 = (__builtin_bit_cast(v2h, p1) + off2) * sc2;
            b[4*d + 0] = h0[0];
            b[4*d + 1] = h0[1];
            b[4*d + 2] = h1[0];
            b[4*d + 3] = h1[1];
        }

        acc = __builtin_amdgcn_wmma_f32_16x16x32_f16(
                  /*neg_a=*/false, a, /*neg_b=*/false, b,
                  /*c_mod=*/(short)0, acc, /*reuse_a=*/false, /*reuse_b=*/false);
    }

    // C/D layout: VGPR r -> M = r + hi*8, N = nCol. Split-K partials via f32 atomics.
    const int mBase = hi * 8;
    #pragma unroll
    for (int r = 0; r < 8; ++r)
        atomicAdd(&ws[(size_t)(mBase + r) * NDIM + nCol], acc[r]);
}

// out = fp16(acc) + bias  (fp16 add, matching reference)
__global__ void __launch_bounds__(256)
finalize_bias_kernel(const float* __restrict__ ws,
                     const _Float16* __restrict__ bias,
                     _Float16* __restrict__ out) {
    int i = blockIdx.x * 256 + threadIdx.x;
    _Float16 h = (_Float16)ws[i];
    out[i] = h + bias[i & (NDIM - 1)];
}

extern "C" void kernel_launch(void* const* d_in, const int* in_sizes, int n_in,
                              void* d_out, int out_size, void* d_ws, size_t ws_size,
                              hipStream_t stream) {
    const float*       x      = (const float*)d_in[0];
    const signed char* qw     = (const signed char*)d_in[1];
    const _Float16*    scales = (const _Float16*)d_in[2];
    const _Float16*    bias   = (const _Float16*)d_in[3];
    _Float16*          out    = (_Float16*)d_out;
    float*             ws     = (float*)d_ws;

    // Zero split-K accumulator (graph-capture safe, deterministic per call).
    hipMemsetAsync(ws, 0, (size_t)BATCH * NDIM * sizeof(float), stream);

    dim3 grid(NDIM / 128, NSPLIT);
    int8_gemm_wmma_kernel<<<grid, 256, 0, stream>>>(x, qw, scales, ws);

    finalize_bias_kernel<<<(BATCH * NDIM) / 256, 256, 0, stream>>>(ws, bias, out);
}
